// BasicDecoder_38311108280801
// MI455X (gfx1250) — matrix operand order repaired and checked by reading, compiled, not run on hardware
//
#include <hip/hip_runtime.h>

// ---------------------------------------------------------------------------
// BasicDecoder on MI455X (gfx1250): 2-layer LSTM + greedy decode, T=20 steps.
// bf16 weights converted once (~95MB, L2-resident); all GEMMs via
// v_wmma_f32_16x16x32_bf16 with fused LSTM-cell epilogues. B-fragments are
// shared across M-tiles within a wave to minimize per-step L2 weight traffic.
// ---------------------------------------------------------------------------

#define Bdim 64
#define Edim 512
#define Hdim 1024
#define Vdim 30000
#define Tdim 20
#define START_TOK 1

typedef __attribute__((ext_vector_type(16))) __bf16 v16bf;
typedef __attribute__((ext_vector_type(8)))  __bf16 v8bf;
typedef __attribute__((ext_vector_type(8)))  float  v8f;

__device__ __forceinline__ unsigned short f2bf(float f) {
  unsigned int u = __float_as_uint(f);
  if ((u & 0x7fffffffu) > 0x7f800000u) return (unsigned short)((u >> 16) | 0x40); // NaN
  return (unsigned short)((u + 0x7fffu + ((u >> 16) & 1u)) >> 16);                // RNE
}

__device__ __forceinline__ float sigm(float x) { return 1.0f / (1.0f + __expf(-x)); }

// ---------------- fragment loaders (bf16, wave32 WMMA layouts) ----------------
// A (16x32, MxK): lane 0-15 -> M=lane, K chunks {0..7, 16..23}; lane 16-31 ->
// M=lane-16, K chunks {8..15, 24..31}. Two contiguous 16B loads per lane.
__device__ __forceinline__ v16bf load_a(const unsigned short* A, int lda,
                                        int row, int k0, int half) {
  const unsigned short* p = A + (size_t)row * lda + k0 + half * 8;
  v8bf c0 = *(const v8bf*)(p);
  v8bf c1 = *(const v8bf*)(p + 16);
  v16bf a;
#pragma unroll
  for (int i = 0; i < 8; ++i) { a[i] = c0[i]; a[i + 8] = c1[i]; }
  return a;
}

// B (32x16, KxN): lane 0-15 -> N=lane, K chunks {0..7, 16..23}; lane 16-31 ->
// N=lane-16, K chunks {8..15, 24..31}. Column n of B == row n of row-major W: two 16B loads.
__device__ __forceinline__ v16bf load_b(const unsigned short* W, int ldw,
                                        int nrow, int k0, int half) {
  const unsigned short* p = W + (size_t)nrow * ldw + k0 + half * 8;
  v8bf c0 = *(const v8bf*)(p);
  v8bf c1 = *(const v8bf*)(p + 16);
  v16bf b;
#pragma unroll
  for (int i = 0; i < 8; ++i) { b[i] = c0[i]; b[i + 8] = c1[i]; }
  return b;
}

// Accumulate one K-segment into 2 M-tiles x 4 gate quadrants (B loaded once
// per quadrant, reused for both M-tiles).
template <int KSEG>
__device__ __forceinline__ void gemm_seg2(v8f acc[2][4], const unsigned short* A, int lda,
                                          const unsigned short* W, int ldw, int wkoff,
                                          int m0, int j0, int l15, int half) {
  const int nbase = j0 + l15;
#pragma unroll 4
  for (int kk = 0; kk < KSEG; kk += 32) {
    v16bf a0 = load_a(A, lda, m0 + l15, kk, half);
    v16bf a1 = load_a(A, lda, m0 + 16 + l15, kk, half);
#pragma unroll
    for (int q = 0; q < 4; ++q) {
      v16bf b = load_b(W, ldw, q * Hdim + nbase, wkoff + kk, half);
      acc[0][q] = __builtin_amdgcn_wmma_f32_16x16x32_bf16(false, a0, false, b,
                                                          (short)0, acc[0][q], false, false);
      acc[1][q] = __builtin_amdgcn_wmma_f32_16x16x32_bf16(false, a1, false, b,
                                                          (short)0, acc[1][q], false, false);
    }
  }
}

// Embedding segment for 2 M-tiles: A rows gathered via token ids.
__device__ __forceinline__ void gemm_emb2(v8f acc[2][4], const unsigned short* Wemb,
                                          const int* token, const unsigned short* Wih,
                                          int m0, int j0, int l15, int half) {
  const unsigned short* arow0 = Wemb + (size_t)token[m0 + l15] * Edim;
  const unsigned short* arow1 = Wemb + (size_t)token[m0 + 16 + l15] * Edim;
  const int nbase = j0 + l15;
#pragma unroll 4
  for (int kk = 0; kk < Edim; kk += 32) {
    const unsigned short* p0 = arow0 + kk + half * 8;
    const unsigned short* p1 = arow1 + kk + half * 8;
    v8bf c0 = *(const v8bf*)(p0), c1 = *(const v8bf*)(p0 + 16);
    v8bf d0 = *(const v8bf*)(p1), d1 = *(const v8bf*)(p1 + 16);
    v16bf a0, a1;
#pragma unroll
    for (int i = 0; i < 8; ++i) {
      a0[i] = c0[i]; a0[i + 8] = c1[i];
      a1[i] = d0[i]; a1[i + 8] = d1[i];
    }
#pragma unroll
    for (int q = 0; q < 4; ++q) {
      v16bf b = load_b(Wih, 2 * Edim, q * Hdim + nbase, Edim + kk, half);
      acc[0][q] = __builtin_amdgcn_wmma_f32_16x16x32_bf16(false, a0, false, b,
                                                          (short)0, acc[0][q], false, false);
      acc[1][q] = __builtin_amdgcn_wmma_f32_16x16x32_bf16(false, a1, false, b,
                                                          (short)0, acc[1][q], false, false);
    }
  }
}

// Fused LSTM cell epilogue for one M-tile.
// C/D layout: VGPR r holds M = m0 + 8*half + r, N = j0 + l15.
__device__ __forceinline__ void lstm_epilogue(v8f acc[4], float b0, float b1, float b2,
                                              float b3, float* c, unsigned short* h_out,
                                              int m0, int j, int half) {
#pragma unroll
  for (int r = 0; r < 8; ++r) {
    const int m = m0 + half * 8 + r;
    const size_t idx = (size_t)m * Hdim + j;
    const float ig = acc[0][r] + b0;
    const float fg = acc[1][r] + b1;
    const float gg = acc[2][r] + b2;
    const float og = acc[3][r] + b3;
    const float cn = sigm(fg) * c[idx] + sigm(ig) * tanhf(gg);
    const float hn = sigm(og) * tanhf(cn);
    c[idx] = cn;
    h_out[idx] = f2bf(hn);
  }
}

// ---------------- kernels ----------------

__global__ void k_convert(const float* __restrict__ src, unsigned short* __restrict__ dst,
                          size_t n) {
  size_t i = (size_t)blockIdx.x * blockDim.x + threadIdx.x;
  const size_t stride = (size_t)gridDim.x * blockDim.x;
  for (; i < n; i += stride) dst[i] = f2bf(src[i]);
}

__global__ void k_zero(unsigned int* p, size_t nwords) {
  size_t i = (size_t)blockIdx.x * blockDim.x + threadIdx.x;
  const size_t stride = (size_t)gridDim.x * blockDim.x;
  for (; i < nwords; i += stride) p[i] = 0u;
}

__global__ void k_set_token(int* token) {
  if (threadIdx.x < Bdim) token[threadIdx.x] = START_TOK;
}

__global__ void k_row0(float* __restrict__ out) {
  size_t i = (size_t)blockIdx.x * blockDim.x + threadIdx.x;
  if (i >= (size_t)Bdim * Vdim) return;
  const int b = (int)(i / Vdim);
  const int v = (int)(i % Vdim);
  out[(size_t)b * Tdim * Vdim + v] = (v == START_TOK) ? 1.0f : 0.0f;
}

// Layer-1 LSTM: gates = [enc | emb] @ Wih1^T + h1 @ Whh1^T + biases, then cell.
// 128 waves: 2 M-tile-pairs x 64 J-tiles; each wave owns 2 M-tiles.
__global__ __launch_bounds__(256) void k_lstm1(
    const unsigned short* __restrict__ enc_bf, const unsigned short* __restrict__ Wemb_bf,
    const int* __restrict__ token, int use_emb,
    const unsigned short* __restrict__ Wih1, const unsigned short* __restrict__ Whh1,
    const float* __restrict__ bih, const float* __restrict__ bhh,
    const unsigned short* __restrict__ h_in, float* __restrict__ c,
    unsigned short* __restrict__ h_out) {
  const int lane = threadIdx.x & 31;
  const int wid = blockIdx.x * (blockDim.x >> 5) + (threadIdx.x >> 5);
  const int jt = wid & 63, p = wid >> 6;   // 64 J-tiles x 2 M-pairs
  const int m0 = p * 32, j0 = jt * 16;
  const int l15 = lane & 15, half = lane >> 4;
  const v8f z = {0.f, 0.f, 0.f, 0.f, 0.f, 0.f, 0.f, 0.f};
  v8f acc[2][4] = {{z, z, z, z}, {z, z, z, z}};

  gemm_seg2<Edim>(acc, enc_bf, Edim, Wih1, 2 * Edim, 0, m0, j0, l15, half);
  if (use_emb) gemm_emb2(acc, Wemb_bf, token, Wih1, m0, j0, l15, half);
  gemm_seg2<Hdim>(acc, h_in, Hdim, Whh1, Hdim, 0, m0, j0, l15, half);

  const int j = j0 + l15;
  const float b0 = bih[j]            + bhh[j];
  const float b1 = bih[Hdim + j]     + bhh[Hdim + j];
  const float b2 = bih[2 * Hdim + j] + bhh[2 * Hdim + j];
  const float b3 = bih[3 * Hdim + j] + bhh[3 * Hdim + j];
  lstm_epilogue(acc[0], b0, b1, b2, b3, c, h_out, m0,      j, half);
  lstm_epilogue(acc[1], b0, b1, b2, b3, c, h_out, m0 + 16, j, half);
}

// Layer-2 LSTM: gates = h1 @ Wih2^T + h2 @ Whh2^T + biases, then cell.
__global__ __launch_bounds__(256) void k_lstm2(
    const unsigned short* __restrict__ h1, const unsigned short* __restrict__ Wih2,
    const unsigned short* __restrict__ Whh2,
    const float* __restrict__ bih, const float* __restrict__ bhh,
    const unsigned short* __restrict__ h_in, float* __restrict__ c,
    unsigned short* __restrict__ h_out) {
  const int lane = threadIdx.x & 31;
  const int wid = blockIdx.x * (blockDim.x >> 5) + (threadIdx.x >> 5);
  const int jt = wid & 63, p = wid >> 6;
  const int m0 = p * 32, j0 = jt * 16;
  const int l15 = lane & 15, half = lane >> 4;
  const v8f z = {0.f, 0.f, 0.f, 0.f, 0.f, 0.f, 0.f, 0.f};
  v8f acc[2][4] = {{z, z, z, z}, {z, z, z, z}};

  gemm_seg2<Hdim>(acc, h1, Hdim, Wih2, Hdim, 0, m0, j0, l15, half);
  gemm_seg2<Hdim>(acc, h_in, Hdim, Whh2, Hdim, 0, m0, j0, l15, half);

  const int j = j0 + l15;
  const float b0 = bih[j]            + bhh[j];
  const float b1 = bih[Hdim + j]     + bhh[Hdim + j];
  const float b2 = bih[2 * Hdim + j] + bhh[2 * Hdim + j];
  const float b3 = bih[3 * Hdim + j] + bhh[3 * Hdim + j];
  lstm_epilogue(acc[0], b0, b1, b2, b3, c, h_out, m0,      j, half);
  lstm_epilogue(acc[1], b0, b1, b2, b3, c, h_out, m0 + 16, j, half);
}

// logits[64,30000] = h2 @ Wout^T + bout. One wave per V-tile; each B fragment
// (Wout from L2) is loaded ONCE and reused across all 4 M-tiles (A from L0).
__global__ __launch_bounds__(256) void k_logits(
    const unsigned short* __restrict__ h2, const unsigned short* __restrict__ Wout_bf,
    const float* __restrict__ bout, float* __restrict__ logits) {
  const int VT = Vdim / 16;  // 1875
  const int lane = threadIdx.x & 31;
  const int wid = blockIdx.x * (blockDim.x >> 5) + (threadIdx.x >> 5);
  if (wid >= VT) return;
  const int v0 = wid * 16;
  const int l15 = lane & 15, half = lane >> 4;
  const v8f z = {0.f, 0.f, 0.f, 0.f, 0.f, 0.f, 0.f, 0.f};
  v8f acc[4] = {z, z, z, z};
  const int nrow = v0 + l15;
#pragma unroll 4
  for (int kk = 0; kk < Hdim; kk += 32) {
    v16bf b = load_b(Wout_bf, Hdim, nrow, kk, half);
#pragma unroll
    for (int mt = 0; mt < 4; ++mt) {
      v16bf a = load_a(h2, Hdim, mt * 16 + l15, kk, half);
      acc[mt] = __builtin_amdgcn_wmma_f32_16x16x32_bf16(false, a, false, b,
                                                        (short)0, acc[mt], false, false);
    }
  }
  const int v = v0 + l15;
  const float bb = bout[v];
#pragma unroll
  for (int mt = 0; mt < 4; ++mt)
#pragma unroll
    for (int r = 0; r < 8; ++r)
      logits[(size_t)(mt * 16 + half * 8 + r) * Vdim + v] = acc[mt][r] + bb;
}

// Per-row log_softmax + argmax; writes output row and next token.
__global__ __launch_bounds__(256) void k_softmax(const float* __restrict__ logits,
                                                 float* __restrict__ out_t,
                                                 int* __restrict__ token) {
  __shared__ float smax[256];
  __shared__ int   sidx[256];
  __shared__ float ssum[256];
  const int b = blockIdx.x;
  const float* row = logits + (size_t)b * Vdim;

  float m = -3.0e38f; int mi = 0;
  for (int v = threadIdx.x; v < Vdim; v += blockDim.x) {
    const float x = row[v];
    if (x > m) { m = x; mi = v; }
  }
  smax[threadIdx.x] = m; sidx[threadIdx.x] = mi;
  __syncthreads();
  for (int s = 128; s > 0; s >>= 1) {
    if ((int)threadIdx.x < s) {
      const float ox = smax[threadIdx.x + s]; const int oi = sidx[threadIdx.x + s];
      if (ox > smax[threadIdx.x] ||
          (ox == smax[threadIdx.x] && oi < sidx[threadIdx.x])) {
        smax[threadIdx.x] = ox; sidx[threadIdx.x] = oi;
      }
    }
    __syncthreads();
  }
  const float gmax = smax[0];

  float s = 0.f;
  for (int v = threadIdx.x; v < Vdim; v += blockDim.x) s += __expf(row[v] - gmax);
  ssum[threadIdx.x] = s;
  __syncthreads();
  for (int st = 128; st > 0; st >>= 1) {
    if ((int)threadIdx.x < st) ssum[threadIdx.x] += ssum[threadIdx.x + st];
    __syncthreads();
  }
  const float lse = gmax + __logf(ssum[0]);

  float* orow = out_t + (size_t)b * Tdim * Vdim;
  for (int v = threadIdx.x; v < Vdim; v += blockDim.x) orow[v] = row[v] - lse;
  if (threadIdx.x == 0) token[b] = sidx[0];
}

// ---------------- host ----------------

extern "C" void kernel_launch(void* const* d_in, const int* in_sizes, int n_in,
                              void* d_out, int out_size, void* d_ws, size_t ws_size,
                              hipStream_t stream) {
  (void)in_sizes; (void)n_in; (void)out_size; (void)ws_size;
  const float* enc  = (const float*)d_in[0];
  const float* Wemb = (const float*)d_in[1];
  const float* Wih1 = (const float*)d_in[2];
  const float* Whh1 = (const float*)d_in[3];
  const float* bih1 = (const float*)d_in[4];
  const float* bhh1 = (const float*)d_in[5];
  const float* Wih2 = (const float*)d_in[6];
  const float* Whh2 = (const float*)d_in[7];
  const float* bih2 = (const float*)d_in[8];
  const float* bhh2 = (const float*)d_in[9];
  const float* Wout = (const float*)d_in[10];
  const float* bout = (const float*)d_in[11];
  float* out = (float*)d_out;

  char* ws = (char*)d_ws;
  size_t off = 0;
  auto carve = [&](size_t bytes) -> char* {
    char* p = ws + off;
    off += (bytes + 255) & ~(size_t)255;
    return p;
  };

  const size_t nWih = (size_t)4 * Hdim * (2 * Edim);  // 4096 x 1024
  const size_t nWhh = (size_t)4 * Hdim * Hdim;        // 4096 x 1024
  unsigned short* Wih1b = (unsigned short*)carve(nWih * 2);
  unsigned short* Whh1b = (unsigned short*)carve(nWhh * 2);
  unsigned short* Wih2b = (unsigned short*)carve(nWhh * 2);
  unsigned short* Whh2b = (unsigned short*)carve(nWhh * 2);
  unsigned short* Woutb = (unsigned short*)carve((size_t)Vdim * Hdim * 2);
  unsigned short* Wembb = (unsigned short*)carve((size_t)Vdim * Edim * 2);
  unsigned short* encb  = (unsigned short*)carve((size_t)Bdim * Edim * 2);
  // state block (zeroed each launch; carved contiguously)
  unsigned short* h1a = (unsigned short*)carve((size_t)Bdim * Hdim * 2);
  unsigned short* h1b = (unsigned short*)carve((size_t)Bdim * Hdim * 2);
  unsigned short* h2a = (unsigned short*)carve((size_t)Bdim * Hdim * 2);
  unsigned short* h2b = (unsigned short*)carve((size_t)Bdim * Hdim * 2);
  float* c1 = (float*)carve((size_t)Bdim * Hdim * 4);
  float* c2 = (float*)carve((size_t)Bdim * Hdim * 4);
  float* logits = (float*)carve((size_t)Bdim * Vdim * 4);
  int* tok = (int*)carve(256);

  // 1) convert weights/activations to bf16 (one pass; thereafter L2-resident)
  k_convert<<<2048, 256, 0, stream>>>(Wih1, Wih1b, nWih);
  k_convert<<<2048, 256, 0, stream>>>(Whh1, Whh1b, nWhh);
  k_convert<<<2048, 256, 0, stream>>>(Wih2, Wih2b, nWhh);
  k_convert<<<2048, 256, 0, stream>>>(Whh2, Whh2b, nWhh);
  k_convert<<<2048, 256, 0, stream>>>(Wout, Woutb, (size_t)Vdim * Hdim);
  k_convert<<<2048, 256, 0, stream>>>(Wemb, Wembb, (size_t)Vdim * Edim);
  k_convert<<<128, 256, 0, stream>>>(enc, encb, (size_t)Bdim * Edim);

  // 2) zero state (h1a..c2 contiguous: 4*128KB bf16 + 2*256KB f32 = 1MB)
  const size_t state_words = ((size_t)Bdim * Hdim * 2 * 4 + (size_t)Bdim * Hdim * 4 * 2) / 4;
  k_zero<<<256, 256, 0, stream>>>((unsigned int*)h1a, state_words);
  k_set_token<<<1, 64, 0, stream>>>(tok);

  unsigned short* h1buf[2] = {h1a, h1b};
  unsigned short* h2buf[2] = {h2a, h2b};
  int cur1 = 0, cur2 = 0;

  // 3) pre-loop step: x0 = [enc | 0], h/c start at zero
  k_lstm1<<<16, 256, 0, stream>>>(encb, Wembb, tok, /*use_emb=*/0, Wih1b, Whh1b,
                                  bih1, bhh1, h1buf[cur1], c1, h1buf[cur1 ^ 1]);
  cur1 ^= 1;
  k_lstm2<<<16, 256, 0, stream>>>(h1buf[cur1], Wih2b, Whh2b, bih2, bhh2,
                                  h2buf[cur2], c2, h2buf[cur2 ^ 1]);
  cur2 ^= 1;

  // 4) output row 0: one-hot(<start>)
  k_row0<<<(Bdim * Vdim + 255) / 256, 256, 0, stream>>>(out);

  // 5) decode steps t = 1..T-1
  for (int t = 1; t < Tdim; ++t) {
    k_lstm1<<<16, 256, 0, stream>>>(encb, Wembb, tok, /*use_emb=*/1, Wih1b, Whh1b,
                                    bih1, bhh1, h1buf[cur1], c1, h1buf[cur1 ^ 1]);
    cur1 ^= 1;
    k_lstm2<<<16, 256, 0, stream>>>(h1buf[cur1], Wih2b, Whh2b, bih2, bhh2,
                                    h2buf[cur2], c2, h2buf[cur2 ^ 1]);
    cur2 ^= 1;
    k_logits<<<(Vdim / 16 + 7) / 8, 256, 0, stream>>>(h2buf[cur2], Woutb, bout, logits);
    k_softmax<<<Bdim, 256, 0, stream>>>(logits, out + (size_t)t * Vdim, tok);
  }
}
